// MoE_10514079941221
// MI455X (gfx1250) — matrix operand order repaired and checked by reading, not compiled, not run on hardware
//
#include <hip/hip_runtime.h>
#include <hip/hip_bf16.h>
#include <math.h>

// ---------------------------------------------------------------------------
// MoE (top-2 of 8 experts), d_model=1024, d_ff=4096, T=4096 tokens.
// Strategy: route first (4x FLOP reduction vs dense reference), convert
// x/W1/W2 to bf16 (8x WMMA math rate vs f32 WMMA; bf16 weights fit in 192MB
// L2), fused per-tile FFN with v_wmma_f32_16x16x32_bf16 and f32 accumulate.
// ---------------------------------------------------------------------------

#define D_MODEL 1024
#define D_FF    4096
#define NEXP    8
#define NTOK    4096            // B*S
#define MT      32              // tokens per FFN block tile
#define XS_STRIDE 1032          // 1024 + 8 bf16 pad  -> row stride 2064B, no bank conflicts
#define AS_STRIDE 136           // 128 + 8 bf16 pad

typedef __bf16        v16bf  __attribute__((ext_vector_type(16)));
typedef float         v8f    __attribute__((ext_vector_type(8)));
typedef unsigned int  u32x4  __attribute__((ext_vector_type(4)));

union Frag { v16bf v; u32x4 q[2]; };

// CDNA5 async global->LDS staging path. The builtin is strictly typed:
//   (int4 addrspace(1)*, int4 addrspace(3)*, imm offset, imm cpol)
#if defined(__AMDGCN__)
# if __has_builtin(__builtin_amdgcn_global_load_async_to_lds_b128) && __has_builtin(__builtin_amdgcn_s_wait_asynccnt)
#  define HAVE_ASYNC_LDS 1
typedef __attribute__((ext_vector_type(4))) int i32x4_t;
typedef __attribute__((address_space(1))) i32x4_t* as1_i32x4_p;
typedef __attribute__((address_space(3))) i32x4_t* as3_i32x4_p;
# endif
#endif
#ifndef HAVE_ASYNC_LDS
# define HAVE_ASYNC_LDS 0
#endif

static __device__ __forceinline__ unsigned short f2bf(float f) {
  unsigned int u = __float_as_uint(f);
  u += 0x7fffu + ((u >> 16) & 1u);        // round-to-nearest-even
  return (unsigned short)(u >> 16);
}

// ---------------------------------------------------------------------------
// 0) zero output + routing counters
// ---------------------------------------------------------------------------
__global__ void moe_zero(float* __restrict__ out, int n, int* __restrict__ cnt) {
  int i = blockIdx.x * blockDim.x + threadIdx.x;
  if (i < NEXP) cnt[i] = 0;
  for (; i < n; i += gridDim.x * blockDim.x) out[i] = 0.0f;
}

// ---------------------------------------------------------------------------
// 1) gate: logits, top-2, softmax over the two scores, bucket scatter
// ---------------------------------------------------------------------------
__global__ void moe_gate(const float* __restrict__ x, const float* __restrict__ Wg,
                         const float* __restrict__ bg, int* __restrict__ cnt,
                         int* __restrict__ bucket, float* __restrict__ bweight) {
  const int t = blockIdx.x * blockDim.x + threadIdx.x;
  if (t >= NTOK) return;
  float acc[NEXP];
#pragma unroll
  for (int e = 0; e < NEXP; ++e) acc[e] = bg[e];
  const float* xr = x + (size_t)t * D_MODEL;
  for (int d = 0; d < D_MODEL; ++d) {
    const float xv = xr[d];
    const float* wr = Wg + d * NEXP;
#pragma unroll
    for (int e = 0; e < NEXP; ++e) acc[e] += xv * wr[e];
  }
  int e0 = 0; float s0 = acc[0];
#pragma unroll
  for (int e = 1; e < NEXP; ++e) if (acc[e] > s0) { s0 = acc[e]; e0 = e; }
  int e1 = (e0 == 0) ? 1 : 0; float s1 = acc[e1];
#pragma unroll
  for (int e = 0; e < NEXP; ++e) if (e != e0 && acc[e] > s1) { s1 = acc[e]; e1 = e; }
  const float w0 = 1.0f / (1.0f + __expf(s1 - s0));   // softmax over {s0, s1}
  const float w1 = 1.0f - w0;
  int slot = atomicAdd(&cnt[e0], 1);
  bucket[e0 * NTOK + slot] = t;  bweight[e0 * NTOK + slot] = w0;
  slot = atomicAdd(&cnt[e1], 1);
  bucket[e1 * NTOK + slot] = t;  bweight[e1 * NTOK + slot] = w1;
}

// ---------------------------------------------------------------------------
// 2) x -> bf16 (elementwise)
// ---------------------------------------------------------------------------
__global__ void conv_x(const float* __restrict__ in, unsigned short* __restrict__ o, int n) {
  for (int i = blockIdx.x * blockDim.x + threadIdx.x; i < n; i += gridDim.x * blockDim.x)
    o[i] = f2bf(in[i]);
}

// ---------------------------------------------------------------------------
// 3) per-expert tiled transpose + f32->bf16:  in [E][R][C] f32 -> out [E][C][R] bf16
//    Gives WMMA B-fragments contiguous-K per lane (b128 loads).
// ---------------------------------------------------------------------------
__global__ void transpose_to_bf16(const float* __restrict__ in,
                                  unsigned short* __restrict__ out, int R, int C) {
  __shared__ float tile[32][33];
  const int e = blockIdx.z;
  const float* ein = in + (size_t)e * R * C;
  unsigned short* eout = out + (size_t)e * R * C;
  const int r0 = blockIdx.y * 32, c0 = blockIdx.x * 32;
  const int lr = threadIdx.x >> 5, lc = threadIdx.x & 31;
#pragma unroll
  for (int i = 0; i < 4; ++i) {
    const int rr = lr + i * 8;
    tile[rr][lc] = ein[(size_t)(r0 + rr) * C + (c0 + lc)];
  }
  __syncthreads();
#pragma unroll
  for (int i = 0; i < 4; ++i) {
    const int cc = lr + i * 8;
    eout[(size_t)(c0 + cc) * R + (r0 + lc)] = f2bf(tile[lc][cc]);
  }
}

// ---------------------------------------------------------------------------
// 4) fused expert FFN: per block = 32 routed tokens of expert e.
//    8 waves; GEMM1 (K=1024) -> GELU/SiLU -> bf16 a-tile in LDS ->
//    GEMM2 accumulates full 32x1024 tile in registers -> weighted atomic add.
// ---------------------------------------------------------------------------
__global__ void __launch_bounds__(256)
moe_ffn(const unsigned short* __restrict__ xbf,   // [T][D] bf16
        const unsigned short* __restrict__ w1t,   // [E][F][D] bf16 (f-major)
        const unsigned short* __restrict__ w2t,   // [E][D][F] bf16 (d-major)
        const float* __restrict__ b1,             // [E][F]
        const float* __restrict__ b2,             // [E][D]
        const int* __restrict__ cnt,
        const int* __restrict__ bucket,
        const float* __restrict__ bw,
        float* __restrict__ out) {
  const int e    = blockIdx.y;
  const int nTok = cnt[e];
  const int m0   = blockIdx.x * MT;
  if (m0 >= nTok) return;

  __shared__ unsigned short xs[MT * XS_STRIDE];   // 32 token rows (bf16)
  __shared__ unsigned short as_[MT * AS_STRIDE];  // activated h chunk (bf16)
  __shared__ int   toks[MT];
  __shared__ float combs[MT];

  const int tid  = threadIdx.x;
  const int wv   = tid >> 5;           // wave 0..7
  const int lane = tid & 31;
  const int ln   = lane & 15;          // N (or M) index within a 16-wide tile
  const int hi   = lane >> 4;          // lane half
  const int kA   = hi << 3;            // A-frag K base (elems): 0|8
  const int kB   = hi << 3;            // B-frag K base (elems): 0|8

  if (tid < MT) {
    const int idx = m0 + tid;
    const int ok  = idx < nTok;
    toks[tid]  = bucket[e * NTOK + (ok ? idx : 0)];
    combs[tid] = ok ? bw[e * NTOK + idx] : 0.0f;   // pad rows contribute 0
  }
  __syncthreads();

  { // stage x rows -> LDS (256B per thread)
    const int row = tid >> 3;
    const int seg = tid & 7;
    const unsigned short* src = xbf + (size_t)toks[row] * D_MODEL + seg * 128;
    unsigned short* dst = &xs[row * XS_STRIDE + seg * 128];
#if HAVE_ASYNC_LDS
#pragma unroll
    for (int i = 0; i < 16; ++i)
      __builtin_amdgcn_global_load_async_to_lds_b128(
          (as1_i32x4_p)(src + i * 8), (as3_i32x4_p)(dst + i * 8), 0, 0);
#else
#pragma unroll
    for (int i = 0; i < 16; ++i)
      *(u32x4*)(dst + i * 8) = *(const u32x4*)(src + i * 8);
#endif
  }
#if HAVE_ASYNC_LDS
  __builtin_amdgcn_s_wait_asynccnt(0);
#endif
  __syncthreads();

  v8f acc[2][8];
#pragma unroll
  for (int mt = 0; mt < 2; ++mt)
#pragma unroll
    for (int nt = 0; nt < 8; ++nt)
#pragma unroll
      for (int j = 0; j < 8; ++j) acc[mt][nt][j] = 0.0f;

  const size_t w1base = (size_t)e * D_FF * D_MODEL;
  const size_t w2base = (size_t)e * D_MODEL * D_FF;
  const bool evenExp = (e & 1) == 0;

  for (int fc = 0; fc < D_FF / 128; ++fc) {
    // ---- GEMM1: h(32 x 16) for this wave's 16 f-columns, K = 1024 ----
    const int fcol = fc * 128 + wv * 16;
    const unsigned short* bp1 = w1t + w1base + (size_t)(fcol + ln) * D_MODEL + kB;
    const float bias1 = b1[e * D_FF + fcol + ln];
    v8f h0, h1;
#pragma unroll
    for (int j = 0; j < 8; ++j) { h0[j] = bias1; h1[j] = bias1; }
    if (fc + 1 < D_FF / 128) __builtin_prefetch(bp1 + (size_t)128 * D_MODEL, 0, 1);
#pragma unroll 4
    for (int kk = 0; kk < D_MODEL / 32; ++kk) {
      const int k0 = kk * 32;
      Frag a0, a1, b;
      b.q[0]  = *(const u32x4*)(bp1 + k0);
      b.q[1]  = *(const u32x4*)(bp1 + k0 + 16);
      a0.q[0] = *(const u32x4*)&xs[ln * XS_STRIDE + k0 + kA];
      a0.q[1] = *(const u32x4*)&xs[ln * XS_STRIDE + k0 + kA + 16];
      a1.q[0] = *(const u32x4*)&xs[(16 + ln) * XS_STRIDE + k0 + kA];
      a1.q[1] = *(const u32x4*)&xs[(16 + ln) * XS_STRIDE + k0 + kA + 16];
      h0 = __builtin_amdgcn_wmma_f32_16x16x32_bf16(false, a0.v, false, b.v,
                                                   (short)0, h0, false, false);
      h1 = __builtin_amdgcn_wmma_f32_16x16x32_bf16(false, a1.v, false, b.v,
                                                   (short)0, h1, false, false);
    }
    // ---- activation (exact-erf GELU for even experts, SiLU for odd) ----
#pragma unroll
    for (int j = 0; j < 8; ++j) {
      const int m = j + (hi << 3);
      const float v0 = h0[j], v1 = h1[j];
      float a0v, a1v;
      if (evenExp) {
        a0v = 0.5f * v0 * (1.0f + erff(v0 * 0.70710678f));
        a1v = 0.5f * v1 * (1.0f + erff(v1 * 0.70710678f));
      } else {
        a0v = v0 / (1.0f + __expf(-v0));
        a1v = v1 / (1.0f + __expf(-v1));
      }
      as_[m * AS_STRIDE + wv * 16 + ln]        = f2bf(a0v);
      as_[(16 + m) * AS_STRIDE + wv * 16 + ln] = f2bf(a1v);
    }
    __syncthreads();
    // ---- GEMM2: acc += a(32 x 128) @ W2t[fchunk][ncols], K = 128 ----
#pragma unroll
    for (int nt = 0; nt < 8; ++nt) {
      const int ncol = wv * 128 + nt * 16 + ln;
      const unsigned short* bp2 = w2t + w2base + (size_t)ncol * D_FF + fc * 128 + kB;
#pragma unroll
      for (int kk = 0; kk < 4; ++kk) {
        const int k0 = kk * 32;
        Frag a0, a1, b;
        b.q[0]  = *(const u32x4*)(bp2 + k0);
        b.q[1]  = *(const u32x4*)(bp2 + k0 + 16);
        a0.q[0] = *(const u32x4*)&as_[ln * AS_STRIDE + k0 + kA];
        a0.q[1] = *(const u32x4*)&as_[ln * AS_STRIDE + k0 + kA + 16];
        a1.q[0] = *(const u32x4*)&as_[(16 + ln) * AS_STRIDE + k0 + kA];
        a1.q[1] = *(const u32x4*)&as_[(16 + ln) * AS_STRIDE + k0 + kA + 16];
        acc[0][nt] = __builtin_amdgcn_wmma_f32_16x16x32_bf16(false, a0.v, false, b.v,
                                                             (short)0, acc[0][nt], false, false);
        acc[1][nt] = __builtin_amdgcn_wmma_f32_16x16x32_bf16(false, a1.v, false, b.v,
                                                             (short)0, acc[1][nt], false, false);
      }
    }
    __syncthreads();
  }

  // ---- epilogue: + b2, x combine weight, atomic accumulate into out ----
#pragma unroll
  for (int nt = 0; nt < 8; ++nt) {
    const int col = wv * 128 + nt * 16 + ln;
    const float bb = b2[e * D_MODEL + col];
#pragma unroll
    for (int j = 0; j < 8; ++j) {
      const int m = j + (hi << 3);
      const float v0 = (acc[0][nt][j] + bb) * combs[m];
      const float v1 = (acc[1][nt][j] + bb) * combs[16 + m];
      atomicAdd(&out[(size_t)toks[m] * D_MODEL + col], v0);
      atomicAdd(&out[(size_t)toks[16 + m] * D_MODEL + col], v1);
    }
  }
}

// ---------------------------------------------------------------------------
// launcher
// ---------------------------------------------------------------------------
extern "C" void kernel_launch(void* const* d_in, const int* in_sizes, int n_in,
                              void* d_out, int out_size, void* d_ws, size_t ws_size,
                              hipStream_t stream) {
  (void)in_sizes; (void)n_in; (void)out_size; (void)ws_size;
  const float* x  = (const float*)d_in[0];
  const float* W1 = (const float*)d_in[1];
  const float* b1 = (const float*)d_in[2];
  const float* W2 = (const float*)d_in[3];
  const float* b2 = (const float*)d_in[4];
  const float* Wg = (const float*)d_in[5];
  const float* bg = (const float*)d_in[6];
  float* out = (float*)d_out;

  // workspace layout (~136.5 MB): counters | buckets | weights | x_bf16 | W1t | W2t
  char* ws = (char*)d_ws;
  int*   cnt    = (int*)ws;
  int*   bucket = (int*)(ws + 256);
  float* bwght  = (float*)(ws + 256 + (size_t)NEXP * NTOK * 4);
  unsigned short* xbf = (unsigned short*)(ws + 256 + 2 * (size_t)NEXP * NTOK * 4);
  unsigned short* w1t = xbf + (size_t)NTOK * D_MODEL;
  unsigned short* w2t = w1t + (size_t)NEXP * D_MODEL * D_FF;

  moe_zero<<<1024, 256, 0, stream>>>(out, NTOK * D_MODEL, cnt);
  moe_gate<<<NTOK / 256, 256, 0, stream>>>(x, Wg, bg, cnt, bucket, bwght);
  conv_x<<<1024, 256, 0, stream>>>(x, xbf, NTOK * D_MODEL);
  transpose_to_bf16<<<dim3(D_FF / 32, D_MODEL / 32, NEXP), 256, 0, stream>>>(W1, w1t, D_MODEL, D_FF);
  transpose_to_bf16<<<dim3(D_MODEL / 32, D_FF / 32, NEXP), 256, 0, stream>>>(W2, w2t, D_FF, D_MODEL);
  moe_ffn<<<dim3(NTOK / MT, NEXP), 256, 0, stream>>>(xbf, w1t, w2t, b1, b2,
                                                     cnt, bucket, bwght, out);
}
